// DynamicConvolution_3710851744253
// MI455X (gfx1250) — matrix operand order repaired and checked by reading, not compiled, not run on hardware
//
#include <hip/hip_runtime.h>

typedef __attribute__((ext_vector_type(16))) __bf16 v16bf;
typedef __attribute__((ext_vector_type(8)))  __bf16 v8bf;
typedef __attribute__((ext_vector_type(8)))  float  v8f;

#define B_N     16
#define CIN     128
#define COUT    128
#define KK      8
#define HIDDEN  32
#define NPIX    4096          // 64*64
#define KDIM    1152          // CIN*3*3
#define KSTEPS  36            // 1152/32
#define TILE_N  64
#define LDA     40            // 32 halves + 8 pad

// dynamic-LDS layout (units: halves / unsigned short)
#define A_TILE  (COUT * LDA)                                   // 5120
#define B_TILE  (TILE_N * LDA)                                 // 2560
#define OFF_A(buf, part) (((buf) * 2 + (part)) * A_TILE)       // 0 .. 20480
#define OFF_B(buf, part) (4 * A_TILE + ((buf) * 2 + (part)) * B_TILE)
#define OFF_KMAP (4 * A_TILE + 4 * B_TILE)                     // 30720
#define SMEM_HALVES (OFF_KMAP + KDIM)                          // 31872 -> 63744 B

extern __shared__ unsigned short smem[];

// round-to-nearest-even float -> bf16 bits
__device__ __forceinline__ unsigned short f2bf(float f) {
  unsigned int u = __float_as_uint(f);
  return (unsigned short)((u + 0x7FFFu + ((u >> 16) & 1u)) >> 16);
}
__device__ __forceinline__ float bf2f(unsigned short h) {
  return __uint_as_float(((unsigned int)h) << 16);
}
// build a v16bf from two aligned 8-half LDS chunks
__device__ __forceinline__ v16bf ld2x8(const unsigned short* p0, const unsigned short* p1) {
  v8bf a = *(const v8bf*)p0;
  v8bf b = *(const v8bf*)p1;
  return __builtin_shufflevector(a, b, 0,1,2,3,4,5,6,7,8,9,10,11,12,13,14,15);
}
// wave-relative LDS byte address (for VDST of global_load_async_to_lds_*)
typedef __attribute__((address_space(3))) const char* lds_cptr_t;
__device__ __forceinline__ unsigned ldsByteOff(const void* p) {
  return (unsigned)(unsigned long long)(lds_cptr_t)p;
}

// ---------------- Phase 1: attention (pool -> MLP -> softmax -> agg bias) ----
__global__ void dynconv_attn(const float* __restrict__ x,
                             const float* __restrict__ w1, const float* __restrict__ b1,
                             const float* __restrict__ w2, const float* __restrict__ b2,
                             const float* __restrict__ kbias,
                             const int* __restrict__ temp,
                             float* __restrict__ alphas, float* __restrict__ aggb) {
  const int b = blockIdx.x, tid = threadIdx.x;
  const int lane = tid & 31, wv = tid >> 5;
  __shared__ float pooled[CIN];
  __shared__ float hbuf[HIDDEN];
  __shared__ float sc[KK];
  __shared__ float sal[KK];

  for (int c = wv; c < CIN; c += 8) {
    const float* row = x + ((size_t)(b * CIN + c) << 12);
    float s = 0.f;
    for (int i = lane; i < NPIX; i += 32) s += row[i];
    #pragma unroll
    for (int off = 16; off > 0; off >>= 1) s += __shfl_down(s, off, 32);
    if (lane == 0) pooled[c] = s * (1.0f / NPIX);
  }
  __syncthreads();
  if (tid < HIDDEN) {
    float a = b1[tid];
    #pragma unroll 4
    for (int c = 0; c < CIN; ++c) a += pooled[c] * w1[tid * CIN + c];
    hbuf[tid] = fmaxf(a, 0.f);
  }
  __syncthreads();
  if (tid < KK) {
    float a = b2[tid];
    #pragma unroll
    for (int j = 0; j < HIDDEN; ++j) a += hbuf[j] * w2[tid * HIDDEN + j];
    sc[tid] = a / (float)temp[0];
  }
  __syncthreads();
  if (tid == 0) {
    float mx = sc[0];
    for (int k = 1; k < KK; ++k) mx = fmaxf(mx, sc[k]);
    float s = 0.f;
    for (int k = 0; k < KK; ++k) { float e = __expf(sc[k] - mx); sal[k] = e; s += e; }
    float inv = 1.0f / s;
    for (int k = 0; k < KK; ++k) sal[k] *= inv;
  }
  __syncthreads();
  if (tid < KK) alphas[b * KK + tid] = sal[tid];
  if (tid < COUT) {
    float s = 0.f;
    #pragma unroll
    for (int k = 0; k < KK; ++k) s += sal[k] * kbias[k * COUT + tid];
    aggb[b * COUT + tid] = s;
  }
}

// ---------------- Phase 2: blend kernel banks, split fp32 -> bf16 hi/lo -----
__global__ void dynconv_aggw(const float* __restrict__ kw,
                             const float* __restrict__ alphas,
                             unsigned short* __restrict__ Whi,
                             unsigned short* __restrict__ Wlo) {
  const int PER_B = COUT * KDIM;        // 147456
  int gid = blockIdx.x * blockDim.x + threadIdx.x;
  if (gid >= B_N * PER_B) return;
  int b = gid / PER_B;
  int r = gid - b * PER_B;              // cout*1152 + cin*9 + kh*3 + kw
  float v = 0.f;
  #pragma unroll
  for (int k = 0; k < KK; ++k)
    v += alphas[b * KK + k] * kw[k * PER_B + r];
  unsigned short hi = f2bf(v);
  Whi[gid] = hi;
  Wlo[gid] = f2bf(v - bf2f(hi));
}

// ---------------- Phase 3 helpers ------------------------------------------
// Async-DMA the A tile (128 couts x 32 K halves, hi+lo) into LDS.
__device__ __forceinline__ void loadA_async(const unsigned short* wh,
                                            const unsigned short* wl,
                                            int kb, int buf, int tid) {
  #pragma unroll
  for (int c = tid; c < 512; c += 256) {
    int row = c >> 2, q = c & 3;
    int src = row * KDIM + kb * 32 + q * 8;
    int dst = row * LDA + q * 8;
    unsigned dh = ldsByteOff(&smem[OFF_A(buf, 0) + dst]);
    unsigned dl = ldsByteOff(&smem[OFF_A(buf, 1) + dst]);
    unsigned long long gh = (unsigned long long)(const void*)(wh + src);
    unsigned long long gl = (unsigned long long)(const void*)(wl + src);
    asm volatile("global_load_async_to_lds_b128 %0, %1, off"
                 :: "v"(dh), "v"(gh) : "memory");
    asm volatile("global_load_async_to_lds_b128 %0, %1, off"
                 :: "v"(dl), "v"(gl) : "memory");
  }
}

// im2col gather of 64 pixels x 32 K, fp32 -> split bf16, into LDS.
__device__ __forceinline__ void loadB_gather(const float* xb, int kb, int buf,
                                             int pixBase, int tid) {
  #pragma unroll
  for (int i = 0; i < 8; ++i) {
    int idx = tid + (i << 8);           // adjacent lanes -> adjacent pixels
    int kk = idx >> 6, n = idx & 63;
    unsigned pk = smem[OFF_KMAP + kb * 32 + kk];
    int cin = pk >> 4, kh = (pk >> 2) & 3, kwd = pk & 3;
    int p  = pixBase + n;
    int ih = (p >> 6) + kh - 1;
    int iw = (p & 63) + kwd - 1;
    float v = 0.f;
    if ((unsigned)ih < 64u && (unsigned)iw < 64u)
      v = xb[((size_t)cin << 12) + (ih << 6) + iw];
    unsigned short hi = f2bf(v);
    smem[OFF_B(buf, 0) + n * LDA + kk] = hi;
    smem[OFF_B(buf, 1) + n * LDA + kk] = f2bf(v - bf2f(hi));
  }
}

// ---------------- Phase 3: implicit-GEMM conv via split-bf16 WMMA -----------
__global__ void __launch_bounds__(256)
dynconv_wmma(const float* __restrict__ x,
             const unsigned short* __restrict__ Whi,
             const unsigned short* __restrict__ Wlo,
             const float* __restrict__ aggb,
             float* __restrict__ out) {
  const int tid  = threadIdx.x;
  const int lane = tid & 31, wv = tid >> 5;
  const int b = blockIdx.y;
  const int pixBase = blockIdx.x * TILE_N;
  const unsigned short* wh = Whi + (size_t)b * COUT * KDIM;
  const unsigned short* wl = Wlo + (size_t)b * COUT * KDIM;
  const float* xb = x + ((size_t)(b * CIN) << 12);

  // one-time (cin,kh,kw) lookup table: kills int-div in the hot gather
  for (int k = tid; k < KDIM; k += 256) {
    int cin = k / 9, rem = k - cin * 9;
    int kh = rem / 3, kwd = rem - kh * 3;
    smem[OFF_KMAP + k] = (unsigned short)((cin << 4) | (kh << 2) | kwd);
  }
  __syncthreads();

  v8f acc[4] = {};

  // prologue: stage K-step 0 into buffer 0
  loadA_async(wh, wl, 0, 0, tid);
  loadB_gather(xb, 0, 0, pixBase, tid);

  for (int kb = 0; kb < KSTEPS; ++kb) {
    const int cur = kb & 1;
    asm volatile("s_wait_asynccnt 0x0" ::: "memory");  // A tile of kb landed
    __syncthreads();                                   // B stores visible; cur^1 free
    if (kb + 1 < KSTEPS) {                             // overlap next-step staging
      loadA_async(wh, wl, kb + 1, cur ^ 1, tid);
      loadB_gather(xb, kb + 1, cur ^ 1, pixBase, tid);
    }

    // --- WMMA: lane layouts per CDNA5 ISA 7.12.2 ---
    // A 16x32 bf16: lane<16 holds K{0..7,16..23}, lane>=16 K{8..15,24..31}
    const int m   = lane & 15;
    const int r0  = (lane >> 4) << 3;   // 0 | 8
    const int kb0 = (lane >> 4) << 3;   // 0 | 8 (B: same K{0..7,16..23}/{8..15,24..31} as A)
    const unsigned short* pah = &smem[OFF_A(cur, 0) + (wv * 16 + m) * LDA + r0];
    const unsigned short* pal = &smem[OFF_A(cur, 1) + (wv * 16 + m) * LDA + r0];
    v16bf ah = ld2x8(pah, pah + 16);
    v16bf al = ld2x8(pal, pal + 16);
    #pragma unroll
    for (int t = 0; t < 4; ++t) {
      const unsigned short* pbh = &smem[OFF_B(cur, 0) + (t * 16 + m) * LDA + kb0];
      const unsigned short* pbl = &smem[OFF_B(cur, 1) + (t * 16 + m) * LDA + kb0];
      v16bf bh = ld2x8(pbh, pbh + 16);
      v16bf bl = ld2x8(pbl, pbl + 16);
      acc[t] = __builtin_amdgcn_wmma_f32_16x16x32_bf16(false, ah, false, bh,
                                                       (short)0, acc[t], false, false);
      acc[t] = __builtin_amdgcn_wmma_f32_16x16x32_bf16(false, ah, false, bl,
                                                       (short)0, acc[t], false, false);
      acc[t] = __builtin_amdgcn_wmma_f32_16x16x32_bf16(false, al, false, bh,
                                                       (short)0, acc[t], false, false);
    }
  }

  // --- store D tiles: VGPR r -> M = r (+8 for upper half-wave), N = lane&15 ---
  const int nlo = lane & 15;
  const int mhi = (lane >> 4) << 3;
  #pragma unroll
  for (int r = 0; r < 8; ++r) {
    int cout = wv * 16 + mhi + r;
    float bias = aggb[b * COUT + cout];
    size_t obase = ((size_t)(b * COUT + cout) << 12) + pixBase + nlo;
    #pragma unroll
    for (int t = 0; t < 4; ++t)
      out[obase + t * 16] = acc[t][r] + bias;
  }
}

// ---------------- host launch ----------------------------------------------
extern "C" void kernel_launch(void* const* d_in, const int* in_sizes, int n_in,
                              void* d_out, int out_size, void* d_ws, size_t ws_size,
                              hipStream_t stream) {
  const float* x     = (const float*)d_in[0];
  const float* kw    = (const float*)d_in[1];
  const float* kbias = (const float*)d_in[2];
  const float* w1    = (const float*)d_in[3];
  const float* b1    = (const float*)d_in[4];
  const float* w2    = (const float*)d_in[5];
  const float* b2    = (const float*)d_in[6];
  const int*   temp  = (const int*)d_in[7];
  float* out = (float*)d_out;

  // workspace layout (all 16B aligned): alphas | aggb | Whi | Wlo
  char* ws = (char*)d_ws;
  float* alphas = (float*)ws;                            // 16*8 f32   (512 B)
  float* aggb   = (float*)(ws + 512);                    // 16*128 f32 (8 KB)
  unsigned short* Whi = (unsigned short*)(ws + 512 + 8192);
  unsigned short* Wlo = Whi + (size_t)B_N * COUT * KDIM; // 4.72 MB each

  dynconv_attn<<<B_N, 256, 0, stream>>>(x, w1, b1, w2, b2, kbias, temp, alphas, aggb);

  const int total = B_N * COUT * KDIM;                   // 2,359,296
  dynconv_aggw<<<(total + 255) / 256, 256, 0, stream>>>(kw, alphas, Whi, Wlo);

  dim3 grid(NPIX / TILE_N, B_N);                         // 64 x 16 workgroups
  dynconv_wmma<<<grid, 256, (unsigned)(SMEM_HALVES * sizeof(unsigned short)),
                 stream>>>(x, Whi, Wlo, aggb, out);
}
